// GPT2StyleAttention_5265629905257
// MI455X (gfx1250) — matrix operand order repaired and checked by reading, not compiled, not run on hardware
//
#include <hip/hip_runtime.h>

// ---------- types ----------
typedef __attribute__((ext_vector_type(16))) __bf16       v16bf;
typedef __attribute__((ext_vector_type(8)))  float        v8f;
typedef __attribute__((ext_vector_type(4)))  unsigned int u32x4;
typedef __attribute__((ext_vector_type(4)))  float        f32x4;
typedef __attribute__((ext_vector_type(8)))  int          i32x8;
typedef __attribute__((ext_vector_type(4)))  int          i32x4;

#if __has_builtin(__builtin_amdgcn_tensor_load_to_lds)
#define HAVE_TDM 1
#else
#define HAVE_TDM 0
#endif

union FragBF {          // one WMMA bf16 operand: 16 bf16 per lane = 8 VGPRs
  u32x4 q[2];
  v16bf v;
};

__device__ __forceinline__ unsigned short f2bf(float f) {
  union { float f; unsigned u; } c; c.f = f;
  unsigned u = c.u;
  unsigned r = (u + 0x7FFFu + ((u >> 16) & 1u)) >> 16;   // round-nearest-even
  return (unsigned short)r;
}

// low 32 bits of a flat shared pointer == LDS byte offset (ISA aperture rule)
__device__ __forceinline__ unsigned lds_off(const void* p) {
  return (unsigned)(unsigned long long)p;
}

__device__ __forceinline__ void wait_tensorcnt0() {
#if __has_builtin(__builtin_amdgcn_s_wait_tensorcnt)
  __builtin_amdgcn_s_wait_tensorcnt(0);
#else
  asm volatile("s_wait_tensorcnt 0x0" ::: "memory");
#endif
}

// =====================================================================
// GEMM: C[M,N] = A[M,K] @ W[K,N] + bias[N]
// AMODE: 0 = A is f32, 1 = A is bf16 (ushort, TDM-staged when available)
// OMODE: 0 = bf16 row-major, 1 = bf16 scattered to [B,H,S,HD], 2 = f32 row-major
// Block tile 128x128, 8 waves (2 M x 4 N), each wave 64x32 via 8 WMMA tiles.
// =====================================================================
template<int AMODE, int OMODE>
__global__ __launch_bounds__(256)
void gemm128_bf16wmma(const void* __restrict__ Ain,
                      const float* __restrict__ W,
                      const float* __restrict__ bias,
                      void* __restrict__ Cout,
                      int M, int N, int K)
{
  __shared__ unsigned short aS[128 * 32];  // [row][k]   bf16
  __shared__ unsigned short wS[128 * 32];  // [n][k]     bf16 (transposed W tile)

  const float*          Af = (const float*)Ain;
  const unsigned short* Ab = (const unsigned short*)Ain;
  unsigned short*       Cb = (unsigned short*)Cout;
  float*                Cf = (float*)Cout;

  const int tid  = threadIdx.x;
  const int lane = tid & 31;
  const int wid  = tid >> 5;
  const int half = lane >> 4;   // lane group (0: lanes 0-15, 1: lanes 16-31)
  const int l16  = lane & 15;
  const int waveM = wid & 1;    // 0..1 -> 64-row slab
  const int waveN = wid >> 1;   // 0..3 -> 32-col slab
  const int blockM = blockIdx.y * 128;
  const int blockN = blockIdx.x * 128;

  v8f acc[4][2] = {};

  for (int k0 = 0; k0 < K; k0 += 32) {
    __syncthreads();                       // LDS reuse guard

    // ---- stage A tile (128x32) as bf16 ----
    bool tdm_done = false;
#if HAVE_TDM
    if (AMODE == 1) {
      tdm_done = true;
      if (tid < 32) {                      // wave 0 drives the Tensor Data Mover
        const unsigned long long ga =
            (unsigned long long)(const void*)(Ab + (size_t)blockM * K + k0);
        u32x4 g0;
        g0[0] = 1u;                                                  // count=1 (valid)
        g0[1] = lds_off(&aS[0]);                                     // lds_addr
        g0[2] = (unsigned)ga;                                        // global_addr[31:0]
        g0[3] = (unsigned)((ga >> 32) & 0x1FFFFFFu) | (2u << 30);    // addr[56:32] | type=2
        i32x8 g1;
        const unsigned d0 = (unsigned)(K - k0);       // tensor_dim0 (remaining cols)
        const unsigned d1 = (unsigned)(M - blockM);   // tensor_dim1 (remaining rows)
        g1[0] = 0x00010000;                           // data_size=2B, no mask/pad/iter
        g1[1] = (int)((d0 & 0xFFFFu) << 16);          // dim0[15:0]   @ bits 63:48
        g1[2] = (int)(((d1 & 0xFFFFu) << 16) | ((d0 >> 16) & 0xFFFFu)); // dim1.lo|dim0.hi
        g1[3] = (int)((32u << 16) | ((d1 >> 16) & 0xFFFFu));            // tile_dim0=32|dim1.hi
        g1[4] = 128;                                  // tile_dim1=128 rows, tile_dim2=0
        g1[5] = K;                                    // tensor_dim0_stride[31:0]
        g1[6] = 0;                                    // stride[47:32] | dim1_stride.lo
        g1[7] = 0;
        i32x4 gz4 = {0, 0, 0, 0};
        i32x8 gz8 = {0, 0, 0, 0, 0, 0, 0, 0};
        __builtin_amdgcn_tensor_load_to_lds(g0, g1, gz4, gz4, gz8, 0);
        wait_tensorcnt0();
      }
    }
#endif
    if (!tdm_done) {
      const int row = tid >> 1;
      const int cb  = (tid & 1) * 16;
      if (AMODE == 0) {
        const float* src = Af + (size_t)(blockM + row) * K + k0 + cb;
        if (k0 + 32 < K) __builtin_prefetch(src + 32, 0, 1);
        #pragma unroll
        for (int v = 0; v < 4; ++v) {
          f32x4 t = ((const f32x4*)src)[v];
          #pragma unroll
          for (int j = 0; j < 4; ++j)
            aS[row * 32 + cb + v * 4 + j] = f2bf(t[j]);
        }
      } else {
        const unsigned short* src = Ab + (size_t)(blockM + row) * K + k0 + cb;
        u32x4 t0 = ((const u32x4*)src)[0];
        u32x4 t1 = ((const u32x4*)src)[1];
        *(u32x4*)&aS[row * 32 + cb]     = t0;
        *(u32x4*)&aS[row * 32 + cb + 8] = t1;
      }
    }

    // ---- stage W tile (32x128) transposed to [n][k] (f32 -> bf16) ----
    {
      const int kk = tid >> 3;
      const int nb = (tid & 7) * 16;
      const float* src = W + (size_t)(k0 + kk) * N + blockN + nb;
      #pragma unroll
      for (int v = 0; v < 4; ++v) {
        f32x4 t = ((const f32x4*)src)[v];
        #pragma unroll
        for (int j = 0; j < 4; ++j)
          wS[(nb + v * 4 + j) * 32 + kk] = f2bf(t[j]);
      }
    }
    __syncthreads();

    // ---- fragments (ISA 16-bit A / B VGPR layouts) ----
    FragBF afr[4];
    #pragma unroll
    for (int mt = 0; mt < 4; ++mt) {
      const unsigned short* base = &aS[(waveM * 64 + mt * 16 + l16) * 32];
      afr[mt].q[0] = *(const u32x4*)(base + half * 8);        // K {0..7}/{8..15}
      afr[mt].q[1] = *(const u32x4*)(base + 16 + half * 8);   // K {16..23}/{24..31}
    }
    FragBF bfr[2];
    #pragma unroll
    for (int nt = 0; nt < 2; ++nt) {
      const unsigned short* base = &wS[(waveN * 32 + nt * 16 + l16) * 32 + half * 8];
      bfr[nt].q[0] = *(const u32x4*)(base);                   // K {0..7}/{8..15}
      bfr[nt].q[1] = *(const u32x4*)(base + 16);
    }
    #pragma unroll
    for (int mt = 0; mt < 4; ++mt)
      #pragma unroll
      for (int nt = 0; nt < 2; ++nt)
        acc[mt][nt] = __builtin_amdgcn_wmma_f32_16x16x32_bf16(
            false, afr[mt].v, false, bfr[nt].v, (short)0, acc[mt][nt], false, false);
  }

  // ---- epilogue: bias + store ----
  #pragma unroll
  for (int mt = 0; mt < 4; ++mt) {
    #pragma unroll
    for (int nt = 0; nt < 2; ++nt) {
      const int n = blockN + waveN * 32 + nt * 16 + l16;
      const float bv = bias[n];
      #pragma unroll
      for (int r = 0; r < 8; ++r) {
        const int m = blockM + waveM * 64 + mt * 16 + r + half * 8;
        const float val = acc[mt][nt][r] + bv;
        if (OMODE == 0) {
          Cb[(size_t)m * N + n] = f2bf(val);
        } else if (OMODE == 1) {               // scatter to [B,H,S,HD] (S=2048,H=16,HD=64)
          const int bidx = m >> 11, s = m & 2047;
          const int hh = n >> 6, hd = n & 63;
          Cb[(((size_t)bidx * 16 + hh) * 2048 + s) * 64 + hd] = f2bf(val);
        } else {
          Cf[(size_t)m * N + n] = val;
        }
      }
    }
  }
}

// =====================================================================
// Causal flash attention: one workgroup per (b, h, 128 q-rows).
// Each wave owns 16 q rows; iterates 32-key chunks with online softmax.
// K tiles staged with async global->LDS copies (ASYNCcnt).
// =====================================================================
__global__ __launch_bounds__(256)
void attn_fwd_wmma(const unsigned short* __restrict__ Qh,
                   const unsigned short* __restrict__ Kh,
                   const unsigned short* __restrict__ Vh,
                   unsigned short* __restrict__ Oc)
{
  constexpr int S = 2048, H = 16, HD = 64;
  __shared__ unsigned short kS[32 * 64];     // [key][d]
  __shared__ unsigned short vS[64 * 32];     // [d][key]  (V^T)
  __shared__ unsigned short pS[8][16 * 32];  // per-wave P tile (row-major)

  const int tid  = threadIdx.x;
  const int lane = tid & 31;
  const int wid  = tid >> 5;
  const int half = lane >> 4;
  const int l16  = lane & 15;
  const int b = blockIdx.z, h = blockIdx.y, qblk = blockIdx.x;
  const int qbase = qblk * 128 + wid * 16;
  const int qb0   = __builtin_amdgcn_readfirstlane(qbase);  // wave-uniform scalar

  const size_t headOff = ((size_t)b * H + h) * S * HD;

  // Q: 16x64 -> two 16x32 A-fragments, loaded once from global
  FragBF qf[2];
  {
    const unsigned short* qrow = Qh + headOff + (size_t)(qbase + l16) * HD;
    #pragma unroll
    for (int f = 0; f < 2; ++f) {
      qf[f].q[0] = *(const u32x4*)(qrow + f * 32 + half * 8);
      qf[f].q[1] = *(const u32x4*)(qrow + f * 32 + 16 + half * 8);
    }
  }

  v8f o[4] = {};
  float m_r[8], l_r[8];
  #pragma unroll
  for (int r = 0; r < 8; ++r) { m_r[r] = -3.0e38f; l_r[r] = 0.0f; }

  const int nch = qblk * 4 + 4;              // key chunks needed under causal mask
  for (int c = 0; c < nch; ++c) {
    const int kbase = c * 32;
    __syncthreads();
    {
      const int key = tid >> 3;
      const int db  = (tid & 7) * 8;
      // K chunk: async global -> LDS (one B128 per lane covers the 4 KB tile)
      {
        const unsigned      ldst  = lds_off(&kS[key * 64 + db]);
        const unsigned long long gsrc =
            (unsigned long long)(const void*)(Kh + headOff + (size_t)(kbase + key) * HD + db);
        asm volatile("global_load_async_to_lds_b128 %0, %1, off"
                     :: "v"(ldst), "v"(gsrc) : "memory");
      }
      // V chunk: manual transpose into vS[d][key] (overlaps with async K copy)
      union { u32x4 v; unsigned short s[8]; } vv;
      vv.v = *(const u32x4*)(Vh + headOff + (size_t)(kbase + key) * HD + db);
      #pragma unroll
      for (int j = 0; j < 8; ++j) vS[(db + j) * 32 + key] = vv.s[j];
      asm volatile("s_wait_asynccnt 0x0" ::: "memory");
    }
    __syncthreads();

    if (kbase <= qb0 + 15) {                 // scalar-uniform causal skip (EXEC all-1)
      v8f s0 = {}, s1 = {};
      #pragma unroll
      for (int f = 0; f < 2; ++f) {
        FragBF kf0, kf1;
        const unsigned short* b0 = &kS[l16 * 64 + f * 32 + half * 8];
        kf0.q[0] = *(const u32x4*)b0;
        kf0.q[1] = *(const u32x4*)(b0 + 16);
        const unsigned short* b1 = &kS[(16 + l16) * 64 + f * 32 + half * 8];
        kf1.q[0] = *(const u32x4*)b1;
        kf1.q[1] = *(const u32x4*)(b1 + 16);
        s0 = __builtin_amdgcn_wmma_f32_16x16x32_bf16(false, qf[f].v, false, kf0.v, (short)0, s0, false, false);
        s1 = __builtin_amdgcn_wmma_f32_16x16x32_bf16(false, qf[f].v, false, kf1.v, (short)0, s1, false, false);
      }
      // scale 1/sqrt(64), causal mask, online softmax (row striped over 16 lanes)
      const int key0 = kbase + l16, key1 = kbase + 16 + l16;
      float alpha[8];
      #pragma unroll
      for (int r = 0; r < 8; ++r) {
        const int qrow = qbase + r + half * 8;
        float a0 = s0[r] * 0.125f; if (key0 > qrow) a0 = -3.0e38f;
        float a1 = s1[r] * 0.125f; if (key1 > qrow) a1 = -3.0e38f;
        float mx = fmaxf(a0, a1);
        #pragma unroll
        for (int d = 1; d < 16; d <<= 1) mx = fmaxf(mx, __shfl_xor(mx, d, 32));
        const float mnew = fmaxf(m_r[r], mx);
        const float p0 = __expf(a0 - mnew);
        const float p1 = __expf(a1 - mnew);
        s0[r] = p0; s1[r] = p1;
        float sum = p0 + p1;
        #pragma unroll
        for (int d = 1; d < 16; d <<= 1) sum += __shfl_xor(sum, d, 32);
        const float al = __expf(m_r[r] - mnew);
        alpha[r] = al;
        l_r[r] = l_r[r] * al + sum;
        m_r[r] = mnew;
      }
      #pragma unroll
      for (int n = 0; n < 4; ++n)
        #pragma unroll
        for (int r = 0; r < 8; ++r) o[n][r] *= alpha[r];

      // P (16x32) through per-wave LDS to reshape into the A-fragment layout
      unsigned short* pw = pS[wid];
      #pragma unroll
      for (int r = 0; r < 8; ++r) {
        const int row = r + half * 8;
        pw[row * 32 + l16]      = f2bf(s0[r]);
        pw[row * 32 + 16 + l16] = f2bf(s1[r]);
      }
      FragBF pf;
      {
        const unsigned short* base = &pw[l16 * 32];
        pf.q[0] = *(const u32x4*)(base + half * 8);
        pf.q[1] = *(const u32x4*)(base + 16 + half * 8);
      }
      #pragma unroll
      for (int n = 0; n < 4; ++n) {
        FragBF vf;
        const unsigned short* base = &vS[(n * 16 + l16) * 32 + half * 8];
        vf.q[0] = *(const u32x4*)base;
        vf.q[1] = *(const u32x4*)(base + 16);
        o[n] = __builtin_amdgcn_wmma_f32_16x16x32_bf16(false, pf.v, false, vf.v, (short)0, o[n], false, false);
      }
    }
  }

  // normalize + store to concat layout [B*S, 1024] (col = h*64 + hd), bf16
  float inv[8];
  #pragma unroll
  for (int r = 0; r < 8; ++r) inv[r] = 1.0f / l_r[r];
  #pragma unroll
  for (int n = 0; n < 4; ++n) {
    const int col = h * 64 + n * 16 + l16;
    #pragma unroll
    for (int r = 0; r < 8; ++r) {
      const int q = qbase + r + half * 8;
      Oc[((size_t)b * S + q) * 1024 + col] = f2bf(o[n][r] * inv[r]);
    }
  }
}

// =====================================================================
// Launch: 3 projection GEMMs -> attention -> output GEMM
// =====================================================================
extern "C" void kernel_launch(void* const* d_in, const int* in_sizes, int n_in,
                              void* d_out, int out_size, void* d_ws, size_t ws_size,
                              hipStream_t stream)
{
  (void)in_sizes; (void)n_in; (void)out_size;
  const float* x  = (const float*)d_in[0];
  const float* Wq = (const float*)d_in[1];
  const float* bq = (const float*)d_in[2];
  const float* Wk = (const float*)d_in[3];
  const float* bk = (const float*)d_in[4];
  const float* Wv = (const float*)d_in[5];
  const float* bv = (const float*)d_in[6];
  const float* Wo = (const float*)d_in[7];
  const float* bo = (const float*)d_in[8];
  float* out = (float*)d_out;

  const int Btot = 4, S = 2048, D = 1024;
  const int M = Btot * S;                                   // 8192
  const size_t headBytes = (size_t)Btot * 16 * S * 64 * sizeof(unsigned short); // 16 MiB

  if (ws_size < 4 * headBytes) return;
  unsigned short* Qh = (unsigned short*)d_ws;
  unsigned short* Kh = (unsigned short*)((char*)d_ws + headBytes);
  unsigned short* Vh = (unsigned short*)((char*)d_ws + 2 * headBytes);
  unsigned short* Oc = (unsigned short*)((char*)d_ws + 3 * headBytes);

  dim3 gg(D / 128, M / 128), bb(256);
  hipLaunchKernelGGL((gemm128_bf16wmma<0, 1>), gg, bb, 0, stream, (const void*)x, Wq, bq, (void*)Qh, M, D, D);
  hipLaunchKernelGGL((gemm128_bf16wmma<0, 1>), gg, bb, 0, stream, (const void*)x, Wk, bk, (void*)Kh, M, D, D);
  hipLaunchKernelGGL((gemm128_bf16wmma<0, 1>), gg, bb, 0, stream, (const void*)x, Wv, bv, (void*)Vh, M, D, D);
  hipLaunchKernelGGL(attn_fwd_wmma, dim3(S / 128, 16, Btot), bb, 0, stream, Qh, Kh, Vh, Oc);
  hipLaunchKernelGGL((gemm128_bf16wmma<1, 2>), gg, bb, 0, stream, (const void*)Oc, Wo, bo, (void*)out, M, D, D);
}
